// GCNResNet_38448547233850
// MI455X (gfx1250) — matrix operand order repaired and checked by reading, not compiled, not run on hardware
//
#include <hip/hip_runtime.h>
#include <hip/hip_bf16.h>
#include <math.h>

// ---------------------------------------------------------------------------
// GCN-ResNet for MI455X (gfx1250, wave32, WMMA + TDM).
// Dense GEMMs -> v_wmma_f32_16x16x32_bf16 (f32 accumulate); W tiles DMA'd to
// LDS by the Tensor Data Mover (tensor_load_to_lds + s_wait_tensorcnt).
// Adjacency: row-sums of A are identically 21, so normalized A == A/21:
//   agg(row i) = (sum_{j in topk(i)} t_j + t_i) / 21        (extra t_i from +I)
// Workspace requirement: ~160 MB.
// ---------------------------------------------------------------------------

typedef __bf16 bf16;
typedef bf16  v16bf __attribute__((ext_vector_type(16)));
typedef bf16  v8bf  __attribute__((ext_vector_type(8)));
typedef bf16  v4bf  __attribute__((ext_vector_type(4)));
typedef float v8f   __attribute__((ext_vector_type(8)));
typedef float v8fl  __attribute__((ext_vector_type(8)));
typedef unsigned int v4u __attribute__((ext_vector_type(4)));
typedef int  v4i __attribute__((ext_vector_type(4)));
typedef int  v8i __attribute__((ext_vector_type(8)));

union ABFrag { v16bf v; v8bf h[2]; };

#define NPTS 1024
#define KNN  20
#define HDIM 256
#define BATCH 16
#define MROWS (BATCH * NPTS)   // 16384

#if __has_builtin(__builtin_amdgcn_tensor_load_to_lds) && \
    __has_builtin(__builtin_amdgcn_s_wait_tensorcnt)
#define GEMM_TDM 1
#endif

// ---------------------------------------------------------------------------
// f32 -> bf16 weight conversion
// ---------------------------------------------------------------------------
__global__ void f32_to_bf16_kernel(const float* __restrict__ in,
                                   bf16* __restrict__ out, int n) {
  int i = blockIdx.x * 256 + threadIdx.x;
  if (i < n) out[i] = (bf16)in[i];
}

// ---------------------------------------------------------------------------
// kNN top-K per row. One block per (b, r): 1024 distances in LDS, then
// 20 argmin passes (tie-break: lower index, matching lax.top_k order).
// ---------------------------------------------------------------------------
__global__ __launch_bounds__(256)
void topk_kernel(const float* __restrict__ x, int* __restrict__ idxOut) {
  const int bn = blockIdx.x;            // b*1024 + r
  const int b  = bn >> 10;
  const int r  = bn & (NPTS - 1);
  __shared__ float d2s[NPTS];
  __shared__ float rv[256];
  __shared__ int   ri[256];
  const float* xb = x + (size_t)b * NPTS * 3;
  const float px = xb[r * 3 + 0], py = xb[r * 3 + 1], pz = xb[r * 3 + 2];
#pragma unroll
  for (int i = 0; i < 4; ++i) {
    int c = threadIdx.x + i * 256;
    float dx = xb[c * 3 + 0] - px;
    float dy = xb[c * 3 + 1] - py;
    float dz = xb[c * 3 + 2] - pz;
    d2s[c] = dx * dx + dy * dy + dz * dz;
  }
  __syncthreads();
  for (int k = 0; k < KNN; ++k) {
    float bv = 3.402823466e38f;
    int   bi = 1 << 30;
#pragma unroll
    for (int i = 0; i < 4; ++i) {
      int c = threadIdx.x + i * 256;
      float v = d2s[c];
      if (v < bv || (v == bv && c < bi)) { bv = v; bi = c; }
    }
    rv[threadIdx.x] = bv; ri[threadIdx.x] = bi;
    __syncthreads();
    for (int s = 128; s > 0; s >>= 1) {
      if (threadIdx.x < s) {
        float ov = rv[threadIdx.x + s]; int oi = ri[threadIdx.x + s];
        if (ov < rv[threadIdx.x] || (ov == rv[threadIdx.x] && oi < ri[threadIdx.x])) {
          rv[threadIdx.x] = ov; ri[threadIdx.x] = oi;
        }
      }
      __syncthreads();
    }
    if (threadIdx.x == 0) {
      idxOut[(size_t)bn * KNN + k] = ri[0];
      d2s[ri[0]] = 3.402823466e38f;   // mask out for next pass
    }
    __syncthreads();
  }
}

// ---------------------------------------------------------------------------
// WMMA GEMM: Y[M,Nout] = X[M,Kin] @ W[Nout,Kin]^T + bias
// Block tile 128x128, 256 threads = 8 wave32s arranged 4(M) x 2(N);
// each wave: 2 x 4 subtiles of 16x16, K stepped by 32 (bf16).
// W tile comes in via the Tensor Data Mover with LDS padding matching KP.
// ---------------------------------------------------------------------------
#define KP 40   // LDS row stride in bf16 elems: 80B rows (16B aligned, bank-shifted)

#ifdef GEMM_TDM
// TDM descriptor: 2D tile [128 rows x 32 bf16], 64B line + 16B pad => 80B rows.
__device__ __forceinline__ void tdm_load_w_tile(const bf16* gsrc, unsigned ldsOff,
                                                int remK, int remN, int ldK) {
  unsigned long long ga = (unsigned long long)(const void*)gsrc;
  v4u g0;
  g0[0] = 1u;                                           // count=1 (valid user D#)
  g0[1] = ldsOff;                                       // lds_addr
  g0[2] = (unsigned)(ga & 0xffffffffu);                 // global_addr lo
  g0[3] = (unsigned)((ga >> 32) & 0x01ffffffu) | (2u << 30);  // addr hi | type=2
  v8i g1;
  g1[0] = (int)((1u << 16)     // data_size = 2 bytes
              | (1u << 20)     // pad_enable
              | (3u << 22)     // pad_interval: 16 DWORDs (64B)
              | (3u << 25));   // pad_amount: 4 DWORDs (16B)
  g1[1] = (int)(((unsigned)remK & 0xffffu) << 16);                       // dim0 lo
  g1[2] = (int)((((unsigned)remK >> 16) & 0xffffu) |
                (((unsigned)remN & 0xffffu) << 16));                     // dim0 hi|dim1 lo
  g1[3] = (int)((((unsigned)remN >> 16) & 0xffffu) | (32u << 16));       // dim1 hi|tile_dim0
  g1[4] = 128;                                                           // tile_dim1 (tile_dim2=0)
  g1[5] = ldK;                                                           // dim0_stride lo
  g1[6] = 0;
  g1[7] = 0;
  v4i gz = {0, 0, 0, 0};
#if __clang_major__ <= 22
  __builtin_amdgcn_tensor_load_to_lds(g0, g1, gz, gz, 0);
#else
  v8i gz8 = {0, 0, 0, 0, 0, 0, 0, 0};
  __builtin_amdgcn_tensor_load_to_lds(g0, g1, gz, gz, gz8, 0);
#endif
}
#endif

__global__ __launch_bounds__(256)
void gemm_bf16_wmma(const float* __restrict__ X, const bf16* __restrict__ Wb,
                    const float* __restrict__ bias, float* __restrict__ Y,
                    int M, int Kin, int Nout) {
  __shared__ bf16 Xs[128 * KP];
  __shared__ bf16 Ws[128 * KP];
  const int tid   = threadIdx.x;
  const int lane  = tid & 31;
  const int wid   = tid >> 5;
  const int waveM = wid & 3;     // 4 waves along M (32 rows each)
  const int waveN = wid >> 2;    // 2 waves along N (64 cols each)
  const int m0 = blockIdx.y * 128;
  const int n0 = blockIdx.x * 128;

  v8f acc[2][4];
#pragma unroll
  for (int mi = 0; mi < 2; ++mi)
#pragma unroll
    for (int ni = 0; ni < 4; ++ni)
#pragma unroll
      for (int e = 0; e < 8; ++e) acc[mi][ni][e] = 0.0f;

  const int nKT = (Kin + 31) >> 5;
  for (int kt = 0; kt < nKT; ++kt) {
    const int k0 = kt * 32;
    const bool full = ((Kin & 7) == 0) && (k0 + 32 <= Kin);  // uniform
    __syncthreads();   // previous iteration's fragment reads are done

#ifdef GEMM_TDM
    // Kick the W-tile DMA first so it overlaps the X conversion below.
    if (wid == 0)
      tdm_load_w_tile(Wb + (size_t)n0 * Kin + k0,
                      (unsigned)(size_t)(void*)&Ws[0],
                      Kin - k0, Nout - n0, Kin);
#endif

    // ---- stage X tile (f32 -> bf16) : 128 rows x 32 cols ----
    if (full) {
#pragma unroll
      for (int i = 0; i < 2; ++i) {
        int li  = tid + i * 256;            // 0..511
        int row = li >> 2;
        int c8  = (li & 3) * 8;
        const float* src = X + (size_t)(m0 + row) * Kin + k0 + c8;
        v8fl f = *(const v8fl*)src;         // 32B vector load
        v8bf t;
#pragma unroll
        for (int j = 0; j < 8; ++j) t[j] = (bf16)f[j];
        *(v8bf*)&Xs[row * KP + c8] = t;     // 16B LDS store
      }
    } else {  // ragged K tail (init layer Kin=3 only)
#pragma unroll
      for (int i = 0; i < 4; ++i) {
        int li  = tid + i * 256;
        int row = li >> 3;
        int c4  = (li & 7) * 4;
        const float* src = X + (size_t)(m0 + row) * Kin + k0 + c4;
        v4bf t;
#pragma unroll
        for (int j = 0; j < 4; ++j)
          t[j] = (k0 + c4 + j < Kin) ? (bf16)src[j] : (bf16)0.0f;
        *(v4bf*)&Xs[row * KP + c4] = t;
      }
    }

#ifdef GEMM_TDM
    if (wid == 0) __builtin_amdgcn_s_wait_tensorcnt(0);
#else
    // ---- stage W tile (already bf16) : 128 rows(N) x 32 cols(K) ----
    if (full) {
#pragma unroll
      for (int i = 0; i < 2; ++i) {
        int li  = tid + i * 256;
        int row = li >> 2;
        int c8  = (li & 3) * 8;
        *(v8bf*)&Ws[row * KP + c8] =
            *(const v8bf*)(Wb + (size_t)(n0 + row) * Kin + k0 + c8);
      }
    } else {
#pragma unroll
      for (int i = 0; i < 4; ++i) {
        int li  = tid + i * 256;
        int row = li >> 3;
        int c4  = (li & 7) * 4;
        const bf16* src = Wb + (size_t)(n0 + row) * Kin + k0 + c4;
        v4bf t;
#pragma unroll
        for (int j = 0; j < 4; ++j)
          t[j] = (k0 + c4 + j < Kin) ? src[j] : (bf16)0.0f;
        *(v4bf*)&Ws[row * KP + c4] = t;
      }
    }
#endif
    __syncthreads();

    // prefetch next K tile of X into cache (global_prefetch_b8)
    if (kt + 1 < nKT) {
      int row = tid >> 1;
      __builtin_prefetch(X + (size_t)(m0 + row) * Kin + k0 + 32, 0, 1);
    }

    // ---- fragment loads (ISA 7.12.2 layouts) ----
    const int fm = lane & 15;      // A: lane -> M row, B: lane -> N col
    const int hi = lane >> 4;      // lane half selects K range
    ABFrag a[2], bb[4];
#pragma unroll
    for (int mi = 0; mi < 2; ++mi) {
      int base = (waveM * 32 + mi * 16 + fm) * KP;
      a[mi].h[0] = *(const v8bf*)&Xs[base + hi * 8];        // K = hi*8 .. +7
      a[mi].h[1] = *(const v8bf*)&Xs[base + 16 + hi * 8];   // K = 16+hi*8 .. +7
    }
#pragma unroll
    for (int ni = 0; ni < 4; ++ni) {
      int base = (waveN * 64 + ni * 16 + fm) * KP; // K = hi*8 .. +7, 16+hi*8 .. +7
      bb[ni].h[0] = *(const v8bf*)&Ws[base + hi * 8];
      bb[ni].h[1] = *(const v8bf*)&Ws[base + 16 + hi * 8];
    }
    // ---- 8 WMMAs per wave per K tile ----
#pragma unroll
    for (int mi = 0; mi < 2; ++mi)
#pragma unroll
      for (int ni = 0; ni < 4; ++ni)
        acc[mi][ni] = __builtin_amdgcn_wmma_f32_16x16x32_bf16(
            false, a[mi].v, false, bb[ni].v, (short)0, acc[mi][ni],
            false, false);
  }

  // ---- epilogue: +bias, store f32 (C/D layout: lane 0-15 M=vgpr, 16-31 M=vgpr+8)
  const int fn = lane & 15;
  const int hi = lane >> 4;
#pragma unroll
  for (int mi = 0; mi < 2; ++mi) {
    int rbase = m0 + waveM * 32 + mi * 16 + hi * 8;
#pragma unroll
    for (int ni = 0; ni < 4; ++ni) {
      int col = n0 + waveN * 64 + ni * 16 + fn;
      float bv = bias ? bias[col] : 0.0f;
#pragma unroll
      for (int r2 = 0; r2 < 8; ++r2)
        Y[(size_t)(rbase + r2) * Nout + col] = acc[mi][ni][r2] + bv;
    }
  }
}

// ---------------------------------------------------------------------------
// Sparse aggregation + ReLU:  y[b,r,:] = relu((sum_k t[b,idx[r,k],:] + t[b,r,:])/21)
// One block per (b,r), 256 threads = channels (coalesced across lanes).
// ---------------------------------------------------------------------------
__global__ __launch_bounds__(256)
void spmm_relu_kernel(const float* __restrict__ T, const int* __restrict__ idx,
                      float* __restrict__ Y) {
  const int bn = blockIdx.x;
  __shared__ int nb[KNN];
  if (threadIdx.x < KNN) nb[threadIdx.x] = idx[(size_t)bn * KNN + threadIdx.x];
  __syncthreads();
  const int rowBase = (bn >> 10) << 10;   // b * N
  const int c = threadIdx.x;
  float acc = T[(size_t)bn * HDIM + c];   // extra self term from +I
#pragma unroll
  for (int k = 0; k < KNN; ++k)
    acc += T[(size_t)(rowBase + nb[k]) * HDIM + c];
  Y[(size_t)bn * HDIM + c] = fmaxf(acc * (1.0f / 21.0f), 0.0f);
}

// ---------------------------------------------------------------------------
// BatchNorm stats (training mode, biased var). 32 channels per block,
// 8 row partitions -> coalesced 128B row segments.
// ---------------------------------------------------------------------------
__global__ __launch_bounds__(256)
void bn_stats_kernel(const float* __restrict__ X, int M, int C,
                     float* __restrict__ mean, float* __restrict__ rsig) {
  const int cl   = threadIdx.x & 31;
  const int part = threadIdx.x >> 5;     // 8 partitions
  const int c    = blockIdx.x * 32 + cl;
  float s = 0.0f, q = 0.0f;
  for (int r = part; r < M; r += 8) {
    float v = X[(size_t)r * C + c];
    s += v; q += v * v;
  }
  __shared__ float ss[256], sq[256];
  ss[threadIdx.x] = s; sq[threadIdx.x] = q;
  __syncthreads();
  if (part == 0) {
#pragma unroll
    for (int p = 1; p < 8; ++p) { s += ss[cl + p * 32]; q += sq[cl + p * 32]; }
    float m = s / (float)M;
    float v = q / (float)M - m * m;
    mean[c] = m;
    rsig[c] = rsqrtf(v + 1e-5f);
  }
}

__global__ __launch_bounds__(256)
void bn_apply_kernel(const float* __restrict__ X, float* __restrict__ Y,
                     const float* __restrict__ mean, const float* __restrict__ rsig,
                     const float* __restrict__ g, const float* __restrict__ bt,
                     const float* __restrict__ res, int n, int C, int doRelu) {
  int i = blockIdx.x * 256 + threadIdx.x;
  if (i >= n) return;
  int c = i % C;   // C is a power of two -> compiles to AND
  float v = (X[i] - mean[c]) * rsig[c] * g[c] + bt[c];
  if (res) v += res[i];
  if (doRelu) v = fmaxf(v, 0.0f);
  Y[i] = v;
}

// ---------------------------------------------------------------------------
// Mean pool over points: [B, N, 1024] -> [B, 1024]
// ---------------------------------------------------------------------------
__global__ __launch_bounds__(256)
void pool_kernel(const float* __restrict__ X, float* __restrict__ Y) {
  int i = blockIdx.x * 256 + threadIdx.x;    // B*1024 threads
  int b = i >> 10, c = i & 1023;
  float s = 0.0f;
  for (int n = 0; n < NPTS; ++n)
    s += X[((size_t)(b << 10) + n) * 1024 + c];
  Y[i] = s * (1.0f / 1024.0f);
}

// ---------------------------------------------------------------------------
// Small dense layers on [16, C] (launch-latency bound; plain VALU is right).
// ---------------------------------------------------------------------------
__global__ __launch_bounds__(256)
void fc_small_kernel(const float* __restrict__ X, const float* __restrict__ W,
                     const float* __restrict__ b, float* __restrict__ Y,
                     int Mr, int Kin, int Nout) {
  int i = blockIdx.x * 256 + threadIdx.x;
  if (i >= Mr * Nout) return;
  int m = i / Nout, n = i % Nout;
  const float* xr = X + (size_t)m * Kin;
  const float* wr = W + (size_t)n * Kin;
  float s = b[n];
  for (int k = 0; k < Kin; ++k) s += xr[k] * wr[k];
  Y[i] = s;
}

__global__ __launch_bounds__(256)
void bn_small_kernel(float* __restrict__ X, const float* __restrict__ g,
                     const float* __restrict__ bt, int C, int doRelu) {
  int c = blockIdx.x * 256 + threadIdx.x;
  if (c >= C) return;
  float s = 0.0f, q = 0.0f;
  for (int r = 0; r < BATCH; ++r) {
    float v = X[(size_t)r * C + c];
    s += v; q += v * v;
  }
  float m  = s * (1.0f / BATCH);
  float va = q * (1.0f / BATCH) - m * m;
  float rs = rsqrtf(va + 1e-5f);
  for (int r = 0; r < BATCH; ++r) {
    float v = (X[(size_t)r * C + c] - m) * rs * g[c] + bt[c];
    X[(size_t)r * C + c] = doRelu ? fmaxf(v, 0.0f) : v;
  }
}

// ---------------------------------------------------------------------------
// Host-side orchestration
// ---------------------------------------------------------------------------
extern "C" void kernel_launch(void* const* d_in, const int* in_sizes, int n_in,
                              void* d_out, int out_size, void* d_ws, size_t ws_size,
                              hipStream_t stream) {
  (void)in_sizes; (void)n_in; (void)out_size; (void)ws_size;

  // Input layout (setup_inputs flattening order):
  const float* x       = (const float*)d_in[0];     // [16,1024,3]
  const float* w_init  = (const float*)d_in[1];     // [256,3]
  const float* b_init  = (const float*)d_in[2];
  const float* bn_g    = (const float*)d_in[3];
  const float* bn_bt   = (const float*)d_in[4];
  // blocks: base 5 + 6*blk: l1.W, l1.b, l2.W, l2.b, bn.g, bn.bt
  const float* w2  = (const float*)d_in[29]; const float* b2  = (const float*)d_in[30];
  const float* g2  = (const float*)d_in[31]; const float* bt2 = (const float*)d_in[32];
  const float* w3  = (const float*)d_in[33]; const float* b3  = (const float*)d_in[34];
  const float* g3  = (const float*)d_in[35]; const float* bt3 = (const float*)d_in[36];
  const float* w4  = (const float*)d_in[37]; const float* b4  = (const float*)d_in[38];
  const float* g4  = (const float*)d_in[39]; const float* bt4 = (const float*)d_in[40];
  const float* w5  = (const float*)d_in[41]; const float* b5  = (const float*)d_in[42];
  const float* g5  = (const float*)d_in[43]; const float* bt5 = (const float*)d_in[44];
  const float* w6  = (const float*)d_in[45]; const float* b6  = (const float*)d_in[46];
  const float* g6  = (const float*)d_in[47]; const float* bt6 = (const float*)d_in[48];
  const float* w7  = (const float*)d_in[49]; const float* b7  = (const float*)d_in[50];

  // Workspace carve-up (~156 MB total).
  char* ws = (char*)d_ws;
  size_t off = 0;
  auto carve = [&](size_t bytes) -> char* {
    char* p = ws + off;
    off += (bytes + 255) & ~(size_t)255;
    return p;
  };
  int*   idxb  = (int*)  carve((size_t)MROWS * KNN * 4);
  bf16*  wbi   = (bf16*) carve(256 * 3 * 2);                 // init_gcn.W
  bf16*  wbl[4][2];
  for (int bk = 0; bk < 4; ++bk)
    for (int l = 0; l < 2; ++l)
      wbl[bk][l] = (bf16*)carve((size_t)256 * 256 * 2);
  bf16*  wb2   = (bf16*) carve((size_t)128 * 256 * 2);
  bf16*  wb3   = (bf16*) carve((size_t)512 * 128 * 2);
  bf16*  wb4   = (bf16*) carve((size_t)1024 * 512 * 2);
  float* act0  = (float*)carve((size_t)MROWS * HDIM * 4);    // h
  float* act1  = (float*)carve((size_t)MROWS * HDIM * 4);    // t (pre-agg)
  float* act2  = (float*)carve((size_t)MROWS * HDIM * 4);    // h1/h2
  float* z128  = (float*)carve((size_t)MROWS * 128 * 4);
  float* z512  = (float*)carve((size_t)MROWS * 512 * 4);
  float* z1024 = (float*)carve((size_t)MROWS * 1024 * 4);
  float* meanb = (float*)carve(1024 * 4);
  float* rsigb = (float*)carve(1024 * 4);
  float* pooled= (float*)carve((size_t)BATCH * 1024 * 4);
  float* s1    = (float*)carve((size_t)BATCH * 1024 * 4);
  float* s2    = (float*)carve((size_t)BATCH * 512 * 4);

  auto conv = [&](const float* src, bf16* dst, int n) {
    f32_to_bf16_kernel<<<(n + 255) / 256, 256, 0, stream>>>(src, dst, n);
  };
  auto gemm = [&](const float* X, const bf16* Wb, const float* bias, float* Y,
                  int M, int Kin, int Nout) {
    dim3 grid(Nout / 128, M / 128);
    gemm_bf16_wmma<<<grid, 256, 0, stream>>>(X, Wb, bias, Y, M, Kin, Nout);
  };
  auto bnpair = [&](float* Xv, float* Yv, const float* g, const float* bt,
                    const float* res, int M, int C, int relu) {
    bn_stats_kernel<<<C / 32, 256, 0, stream>>>(Xv, M, C, meanb, rsigb);
    int n = M * C;
    bn_apply_kernel<<<(n + 255) / 256, 256, 0, stream>>>(
        Xv, Yv, meanb, rsigb, g, bt, res, n, C, relu);
  };

  // 0) convert GEMM weights to bf16
  conv(w_init, wbi, 256 * 3);
  for (int bk = 0; bk < 4; ++bk) {
    conv((const float*)d_in[5 + 6 * bk + 0], wbl[bk][0], 256 * 256);
    conv((const float*)d_in[5 + 6 * bk + 2], wbl[bk][1], 256 * 256);
  }
  conv(w2, wb2, 128 * 256);
  conv(w3, wb3, 512 * 128);
  conv(w4, wb4, 1024 * 512);

  // 1) kNN graph
  topk_kernel<<<MROWS, 256, 0, stream>>>(x, idxb);

  // 2) init GCN layer: t = x @ Wi^T + b; h = relu(A_hat @ t); h = BN(h)
  gemm(x, wbi, b_init, act1, MROWS, 3, HDIM);
  spmm_relu_kernel<<<MROWS, 256, 0, stream>>>(act1, idxb, act0);
  bnpair(act0, act0, bn_g, bn_bt, nullptr, MROWS, HDIM, /*relu=*/0);

  // 3) residual GCN blocks
  for (int bk = 0; bk < 4; ++bk) {
    const float* lb1 = (const float*)d_in[5 + 6 * bk + 1];
    const float* lb2 = (const float*)d_in[5 + 6 * bk + 3];
    const float* bg  = (const float*)d_in[5 + 6 * bk + 4];
    const float* bbt = (const float*)d_in[5 + 6 * bk + 5];
    gemm(act0, wbl[bk][0], lb1, act1, MROWS, HDIM, HDIM);
    spmm_relu_kernel<<<MROWS, 256, 0, stream>>>(act1, idxb, act2);   // h1
    gemm(act2, wbl[bk][1], lb2, act1, MROWS, HDIM, HDIM);
    spmm_relu_kernel<<<MROWS, 256, 0, stream>>>(act1, idxb, act2);   // h2
    bnpair(act2, act0, bg, bbt, /*res=*/act0, MROWS, HDIM, /*relu=*/1);
  }

  // 4) per-point MLP
  gemm(act0, wb2, b2, z128, MROWS, 256, 128);
  bnpair(z128, z128, g2, bt2, nullptr, MROWS, 128, 1);
  gemm(z128, wb3, b3, z512, MROWS, 128, 512);
  bnpair(z512, z512, g3, bt3, nullptr, MROWS, 512, 1);
  gemm(z512, wb4, b4, z1024, MROWS, 512, 1024);
  bnpair(z1024, z1024, g4, bt4, nullptr, MROWS, 1024, 1);

  // 5) mean pool over points -> [16, 1024]
  pool_kernel<<<(BATCH * 1024) / 256, 256, 0, stream>>>(z1024, pooled);

  // 6) head MLP on [16, C]
  fc_small_kernel<<<(BATCH * 1024 + 255) / 256, 256, 0, stream>>>(
      pooled, w5, b5, s1, BATCH, 1024, 1024);
  bn_small_kernel<<<(1024 + 255) / 256, 256, 0, stream>>>(s1, g5, bt5, 1024, 1);
  fc_small_kernel<<<(BATCH * 512 + 255) / 256, 256, 0, stream>>>(
      s1, w6, b6, s2, BATCH, 1024, 512);
  bn_small_kernel<<<(512 + 255) / 256, 256, 0, stream>>>(s2, g6, bt6, 512, 1);
  fc_small_kernel<<<(BATCH * 40 + 255) / 256, 256, 0, stream>>>(
      s2, w7, b7, (float*)d_out, BATCH, 512, 40);
}
